// AttentionLayer_84542136255217
// MI455X (gfx1250) — matrix operand order repaired and checked by reading, not compiled, not run on hardware
//
#include <hip/hip_runtime.h>
#include <hip/hip_bf16.h>

// ---------------------------------------------------------------------------
// MI455X (gfx1250) multi-head attention:
//   prep:  X -> bf16, W -> transposed bf16 (one-time)
//   GEMM:  QKV projection, bf16 WMMA, TDM (tensor_load_to_lds) double-buffered
//   attn:  flash attention, bf16 WMMA, TDM double-buffered K/V tiles
// ---------------------------------------------------------------------------

typedef __bf16 bf16_t;
typedef __attribute__((ext_vector_type(16))) __bf16 v16bf;
typedef __attribute__((ext_vector_type(8)))  __bf16 v8bf;
typedef __attribute__((ext_vector_type(4)))  __bf16 v4bf;
typedef __attribute__((ext_vector_type(8)))  float  v8f;
typedef __attribute__((ext_vector_type(4)))  float  v4f;
typedef __attribute__((ext_vector_type(4)))  unsigned int u32x4;
typedef __attribute__((ext_vector_type(8)))  int    i32x8;
typedef __attribute__((ext_vector_type(4)))  int    i32x4;

#define SEQ    2048
#define DMODEL 1024
#define NHEAD  16
#define HDIM   128
#define MASK_SCALE (-1.0e9f)

#if defined(__has_builtin)
#  if __has_builtin(__builtin_amdgcn_tensor_load_to_lds)
#    define HAVE_TDM 1
#  endif
#endif
#ifndef HAVE_TDM
#  define HAVE_TDM 0
#endif

static __device__ __forceinline__ v16bf mk16(v8bf lo, v8bf hi) {
  v16bf r;
#pragma unroll
  for (int i = 0; i < 8; ++i) { r[i] = lo[i]; r[i + 8] = hi[i]; }
  return r;
}

static __device__ __forceinline__ v8f zero8() {
  v8f r;
#pragma unroll
  for (int i = 0; i < 8; ++i) r[i] = 0.0f;
  return r;
}

static __device__ __forceinline__ v8f wmma_bf16(v16bf a, v16bf b, v8f c) {
  return __builtin_amdgcn_wmma_f32_16x16x32_bf16(false, a, false, b,
                                                 (short)0, c, false, false);
}

#if HAVE_TDM
// Issue one 2D TDM load (global -> LDS), bf16 elements, optional LDS row pad.
// D# group layout per CDNA5 ISA ch.8.3/8.4. dims/strides in elements.
static __device__ __forceinline__ void tdm_load_2d(
    unsigned lds_addr, const void* gptr,
    unsigned dim0, unsigned dim1, unsigned long long stride0,
    unsigned tile0, unsigned tile1,
    unsigned pad_interval_code, unsigned pad_amount_code) {
  unsigned long long ga = (unsigned long long)(uintptr_t)gptr;
  u32x4 g0;
  g0[0] = 1u;                                     // count=1, user descriptor
  g0[1] = lds_addr;                               // LDS byte address
  g0[2] = (unsigned)(ga & 0xFFFFFFFFull);         // global addr [31:0]
  g0[3] = (unsigned)((ga >> 32) & 0x01FFFFFFull)  // global addr [56:32]
          | (2u << 30);                           // type = 2 ("image")
  i32x8 g1;
  unsigned w0 = (1u << 16)                        // data_size = 1 -> 2 bytes
              | (1u << 20)                        // pad_enable
              | (pad_interval_code << 22)
              | (pad_amount_code << 25);
  g1[0] = (int)w0;
  g1[1] = (int)((dim0 & 0xFFFFu) << 16);          // abar_addr=0 | tensor_dim0.lo
  g1[2] = (int)(((dim0 >> 16) & 0xFFFFu) | ((dim1 & 0xFFFFu) << 16));
  g1[3] = (int)(((dim1 >> 16) & 0xFFFFu) | ((tile0 & 0xFFFFu) << 16));
  g1[4] = (int)(tile1 & 0xFFFFu);                 // tile_dim1 | tile_dim2=0
  g1[5] = (int)(unsigned)(stride0 & 0xFFFFFFFFull);
  g1[6] = (int)(unsigned)((stride0 >> 32) & 0xFFFFull);  // dim1_stride=0
  g1[7] = 0;
  i32x4 z4; z4[0] = z4[1] = z4[2] = z4[3] = 0;
#if __clang_major__ >= 23
  i32x8 z8;
#pragma unroll
  for (int i = 0; i < 8; ++i) z8[i] = 0;
  __builtin_amdgcn_tensor_load_to_lds(g0, g1, z4, z4, z8, 0);
#else
  __builtin_amdgcn_tensor_load_to_lds(g0, g1, z4, z4, 0);
#endif
}
#endif  // HAVE_TDM

// ---------------------------------------------------------------------------
// Prep 1: X fp32 -> bf16 (same layout).  Grid: 2048 x 256, 4 elts/thread.
// ---------------------------------------------------------------------------
__global__ __launch_bounds__(256)
void convert_x_kernel(const float* __restrict__ X, bf16_t* __restrict__ Xbf) {
  size_t i = ((size_t)blockIdx.x * 256 + threadIdx.x) * 4;
  v4f x = *(const v4f*)(X + i);
  v4bf b;
#pragma unroll
  for (int u = 0; u < 4; ++u) b[u] = (__bf16)x[u];
  *(v4bf*)(Xbf + i) = b;
}

// ---------------------------------------------------------------------------
// Prep 2: W{q,k,v}[1024][2048] fp32 -> WT[q][n=2048][k=1024] bf16 (transposed).
// LDS-tiled 64x64, coalesced global reads and writes. Grid: (16,32,3) x 256.
// ---------------------------------------------------------------------------
__global__ __launch_bounds__(256)
void transpose_w_kernel(const float* __restrict__ Wq, const float* __restrict__ Wk,
                        const float* __restrict__ Wv, bf16_t* __restrict__ WT) {
  __shared__ __align__(16) float T[64][68];
  const int kt = blockIdx.x;   // k tile (1024/64)
  const int nt = blockIdx.y;   // n tile (2048/64)
  const int q  = blockIdx.z;
  const float* __restrict__ W = (q == 0) ? Wq : (q == 1) ? Wk : Wv;
  const int tid = threadIdx.x;
#pragma unroll
  for (int i = 0; i < 4; ++i) {
    int r = (tid >> 4) + i * 16, c4 = tid & 15;
    *(v4f*)(&T[r][c4 * 4]) =
        *(const v4f*)(W + (size_t)(kt * 64 + r) * (NHEAD * HDIM) + nt * 64 + c4 * 4);
  }
  __syncthreads();
#pragma unroll
  for (int i = 0; i < 2; ++i) {
    int c = tid + i * 256;
    int nr = c >> 3, piece = c & 7;
    v8bf o;
#pragma unroll
    for (int u = 0; u < 8; ++u) o[u] = (__bf16)T[piece * 8 + u][nr];
    *(v8bf*)(WT + ((size_t)q * 2048 + nt * 64 + nr) * DMODEL + kt * 64 + piece * 8) = o;
  }
}

// ---------------------------------------------------------------------------
// Kernel 3: fused QKV GEMM.  Xbf[2048][1024] x WT[n][k] -> Q/K head-major,
// V transposed.  128x128x32 tiles, double-buffered LDS fed by TDM.
// Grid: (48, 16), 256 threads (8 waves).
// ---------------------------------------------------------------------------
__global__ __launch_bounds__(256)
void qkv_gemm_kernel(const bf16_t* __restrict__ Xbf, const bf16_t* __restrict__ WT,
                     const float* __restrict__ bq, const float* __restrict__ bk,
                     const float* __restrict__ bv,
                     bf16_t* __restrict__ Qb, bf16_t* __restrict__ Kb,
                     bf16_t* __restrict__ Vtb) {
  __shared__ __align__(16) bf16_t As[2][128][40];  // X tile [m][k], +8 pad
  __shared__ __align__(16) bf16_t Bs[2][128][40];  // W tile [n][k], +8 pad

  const int nblk = blockIdx.x;        // 0..47
  const int mblk = blockIdx.y;        // 0..15
  const int qkv  = nblk >> 4;
  const int hidx = nblk & 15;
  const float* __restrict__ bias = (qkv == 0) ? bq : (qkv == 1) ? bk : bv;

  const int tid  = threadIdx.x;
  const int wave = tid >> 5;
  const int lane = tid & 31;
  const int l16  = lane & 15;
  const int half = lane >> 4;

  const bf16_t* __restrict__ Abase = Xbf + (size_t)mblk * 128 * DMODEL;
  const bf16_t* __restrict__ Bbase = WT + ((size_t)qkv * 2048 + hidx * HDIM) * DMODEL;

  v8f acc[8];
#pragma unroll
  for (int j = 0; j < 8; ++j) acc[j] = zero8();

#if HAVE_TDM
  // rows of 32 bf16 = 64B = 16 DW -> pad_interval code 3; pad 16B = 4 DW -> 3
  auto issueAB = [&](int buf, int k0) {
    tdm_load_2d((unsigned)(uintptr_t)&As[buf][0][0], Abase + k0,
                32, 128, DMODEL, 32, 128, 3, 3);
    tdm_load_2d((unsigned)(uintptr_t)&Bs[buf][0][0], Bbase + k0,
                32, 128, DMODEL, 32, 128, 3, 3);
  };
  if (wave == 0) issueAB(0, 0);
#endif

  for (int it = 0; it < DMODEL / 32; ++it) {
    const int buf = it & 1;
#if HAVE_TDM
    if (wave == 0) {
      if (it + 1 < DMODEL / 32) {
        issueAB(buf ^ 1, (it + 1) * 32);
        __builtin_amdgcn_s_wait_tensorcnt(2);  // oldest tile (2 ops) done
      } else {
        __builtin_amdgcn_s_wait_tensorcnt(0);
      }
    }
#else
    const int k0 = it * 32;
#pragma unroll
    for (int i = 0; i < 2; ++i) {   // A: 128x32 bf16, 16B chunks
      int c = tid + i * 256, row = c >> 2, piece = c & 3;
      *(v8bf*)(&As[buf][row][piece * 8]) =
          *(const v8bf*)(Abase + (size_t)row * DMODEL + k0 + piece * 8);
    }
#pragma unroll
    for (int i = 0; i < 2; ++i) {   // B: 128x32 bf16
      int c = tid + i * 256, row = c >> 2, piece = c & 3;
      *(v8bf*)(&Bs[buf][row][piece * 8]) =
          *(const v8bf*)(Bbase + (size_t)row * DMODEL + k0 + piece * 8);
    }
#endif
    __syncthreads();

    // A fragment: ISA 16-bit A layout: K = (e/8)*16 + half*8 + (e%8)
    const int arow = wave * 16 + l16;
    v8bf alo = *(const v8bf*)(&As[buf][arow][half * 8]);
    v8bf ahi = *(const v8bf*)(&As[buf][arow][16 + half * 8]);
    v16bf afrag = mk16(alo, ahi);

#pragma unroll
    for (int j = 0; j < 8; ++j) {   // B fragment: K = (e/8)*16 + half*8 + (e%8)
      const int brow = j * 16 + l16;
      v8bf b0 = *(const v8bf*)(&Bs[buf][brow][half * 8]);
      v8bf b1 = *(const v8bf*)(&Bs[buf][brow][16 + half * 8]);
      acc[j] = wmma_bf16(afrag, mk16(b0, b1), acc[j]);
    }
    __syncthreads();
  }

  // Epilogue: bias, convert, route (C layout: row = half*8 + r, col = l16)
  const int sbase = mblk * 128 + wave * 16 + half * 8;
#pragma unroll
  for (int j = 0; j < 8; ++j) {
    const int d = j * 16 + l16;
    const float bv_ = bias[hidx * HDIM + d];
    if (qkv == 2) {
      v8bf pv;
#pragma unroll
      for (int r = 0; r < 8; ++r) pv[r] = (__bf16)(acc[j][r] + bv_);
      *(v8bf*)(Vtb + ((size_t)hidx * HDIM + d) * SEQ + sbase) = pv;  // V^T
    } else {
      bf16_t* __restrict__ P = (qkv == 0) ? Qb : Kb;
#pragma unroll
      for (int r = 0; r < 8; ++r)
        P[((size_t)hidx * SEQ + (sbase + r)) * HDIM + d] = (__bf16)(acc[j][r] + bv_);
    }
  }
}

// ---------------------------------------------------------------------------
// Kernel 4: flash attention; faithful to reference (scores MULTIPLIED by
// mask matrix (m_i & m_j ? -1e9 : 0), softmax over t, @V).
// K/V tiles double-buffered via TDM.  Grid: (16 s-blocks, 16 heads) x 256.
// ---------------------------------------------------------------------------
__global__ __launch_bounds__(256)
void attn_kernel(const bf16_t* __restrict__ Qb, const bf16_t* __restrict__ Kb,
                 const bf16_t* __restrict__ Vtb,
                 const unsigned char* __restrict__ smask,
                 float* __restrict__ out) {
  __shared__ __align__(16) bf16_t Ks[2][64][136];   // K tile [t][d], +8 pad
  __shared__ __align__(16) bf16_t Vts[2][128][72];  // V tile [d][t], +8 pad
  __shared__ __align__(16) bf16_t Ps[8][16][72];    // per-wave P [s][t], +8 pad

  const int sblk = blockIdx.x;
  const int h    = blockIdx.y;
  const int tid  = threadIdx.x;
  const int wave = tid >> 5;
  const int lane = tid & 31;
  const int l16  = lane & 15;
  const int half = lane >> 4;

  const bf16_t* __restrict__ Qh = Qb + (size_t)h * SEQ * HDIM;
  const bf16_t* __restrict__ Kh = Kb + (size_t)h * SEQ * HDIM;
  const bf16_t* __restrict__ Vh = Vtb + (size_t)h * HDIM * SEQ;

  // Q fragments resident in registers (16 rows x 128, 4 x K=32)
  v16bf qf[4];
  {
    const int row = sblk * 128 + wave * 16 + l16;
#pragma unroll
    for (int kk = 0; kk < 4; ++kk) {
      const bf16_t* p = Qh + (size_t)row * HDIM + kk * 32;
      qf[kk] = mk16(*(const v8bf*)(p + half * 8),
                    *(const v8bf*)(p + 16 + half * 8));
    }
  }

  bool rm[8];
#pragma unroll
  for (int r = 0; r < 8; ++r)
    rm[r] = smask[sblk * 128 + wave * 16 + half * 8 + r] != 0;

  float mrun[8], lrun[8];
  v8f O[8];
#pragma unroll
  for (int r = 0; r < 8; ++r) { mrun[r] = -1.0e30f; lrun[r] = 0.0f; }
#pragma unroll
  for (int j = 0; j < 8; ++j) O[j] = zero8();

  const float inv_sqrt_d = 0.08838834764831845f;  // 1/sqrt(128)
  const int NT = SEQ / 64;

#if HAVE_TDM
  auto issueKV = [&](int buf, int t0) {
    // K: rows 64x128 bf16 (256B = 64 DW -> interval code 5), pad 4 DW -> 3
    tdm_load_2d((unsigned)(uintptr_t)&Ks[buf][0][0], Kh + (size_t)t0 * HDIM,
                128, 64, HDIM, 128, 64, 5, 3);
    // Vt: rows 128x64 bf16 (128B = 32 DW -> interval code 4), pad 4 DW -> 3
    tdm_load_2d((unsigned)(uintptr_t)&Vts[buf][0][0], Vh + t0,
                64, 128, SEQ, 64, 128, 4, 3);
  };
  if (wave == 0) issueKV(0, 0);
#endif

  for (int it = 0; it < NT; ++it) {
    const int buf = it & 1;
    const int t0  = it * 64;
#if HAVE_TDM
    if (wave == 0) {
      if (it + 1 < NT) {
        issueKV(buf ^ 1, (it + 1) * 64);
        __builtin_amdgcn_s_wait_tensorcnt(2);
      } else {
        __builtin_amdgcn_s_wait_tensorcnt(0);
      }
    }
#else
#pragma unroll
    for (int i = 0; i < 4; ++i) {
      int c = tid + i * 256, row = c >> 4, piece = c & 15;
      *(v8bf*)(&Ks[buf][row][piece * 8]) =
          *(const v8bf*)(Kh + (size_t)(t0 + row) * HDIM + piece * 8);
    }
#pragma unroll
    for (int i = 0; i < 4; ++i) {
      int c = tid + i * 256, row = c >> 3, piece = c & 7;
      *(v8bf*)(&Vts[buf][row][piece * 8]) =
          *(const v8bf*)(Vh + (size_t)row * SEQ + t0 + piece * 8);
    }
#endif
    __syncthreads();

    // ---- scores: 4 tiles of 16 t-cols, K=128 each (4 WMMAs) ----
    float sc[4][8];
#pragma unroll
    for (int j = 0; j < 4; ++j) {
      v8f c = zero8();
      const int trow = j * 16 + l16;
#pragma unroll
      for (int kk = 0; kk < 4; ++kk) {
        v8bf b0 = *(const v8bf*)(&Ks[buf][trow][kk * 32 + half * 8]);
        v8bf b1 = *(const v8bf*)(&Ks[buf][trow][kk * 32 + 16 + half * 8]);
        c = wmma_bf16(qf[kk], mk16(b0, b1), c);
      }
      const bool cm = smask[t0 + j * 16 + l16] != 0;
#pragma unroll
      for (int r = 0; r < 8; ++r) {
        float s = c[r] * inv_sqrt_d;
        sc[j][r] = (cm && rm[r]) ? s * MASK_SCALE : 0.0f;  // scores * mask
      }
    }

    // ---- online softmax: row max over 64 new cols ----
    float mnew[8], alpha[8];
#pragma unroll
    for (int r = 0; r < 8; ++r) {
      float mx = sc[0][r];
#pragma unroll
      for (int j = 1; j < 4; ++j) mx = fmaxf(mx, sc[j][r]);
      mx = fmaxf(mx, __shfl_xor(mx, 1, 16));
      mx = fmaxf(mx, __shfl_xor(mx, 2, 16));
      mx = fmaxf(mx, __shfl_xor(mx, 4, 16));
      mx = fmaxf(mx, __shfl_xor(mx, 8, 16));
      mnew[r]  = fmaxf(mrun[r], mx);
      alpha[r] = __expf(mrun[r] - mnew[r]);
      mrun[r]  = mnew[r];
    }

    // ---- P = exp(sc - mnew); lane-partial sums; stash to LDS ----
#pragma unroll
    for (int j = 0; j < 4; ++j) {
#pragma unroll
      for (int r = 0; r < 8; ++r) {
        float p = __expf(sc[j][r] - mnew[r]);
        lrun[r] = (j == 0) ? (lrun[r] * alpha[r] + p) : (lrun[r] + p);
        Ps[wave][half * 8 + r][j * 16 + l16] = (__bf16)p;
      }
    }
#pragma unroll
    for (int jj = 0; jj < 8; ++jj)
#pragma unroll
      for (int r = 0; r < 8; ++r) O[jj][r] *= alpha[r];

    asm volatile("s_wait_dscnt 0x0" ::: "memory");  // per-wave P RAW

    // ---- O += P @ V (K=64 as 2 x K=32, 8 dd-tiles) ----
#pragma unroll
    for (int ph = 0; ph < 2; ++ph) {
      const bf16_t* prow = &Ps[wave][l16][ph * 32];
      v16bf pf = mk16(*(const v8bf*)(prow + half * 8),
                      *(const v8bf*)(prow + 16 + half * 8));
#pragma unroll
      for (int jj = 0; jj < 8; ++jj) {
        const int vrow = jj * 16 + l16;
        v8bf b0 = *(const v8bf*)(&Vts[buf][vrow][ph * 32 + half * 8]);
        v8bf b1 = *(const v8bf*)(&Vts[buf][vrow][ph * 32 + 16 + half * 8]);
        O[jj] = wmma_bf16(pf, mk16(b0, b1), O[jj]);
      }
    }
    __syncthreads();  // all readers done before buffer is re-filled
  }

  // ---- finalize ----
#pragma unroll
  for (int r = 0; r < 8; ++r) {
    float l = lrun[r];
    l += __shfl_xor(l, 1, 16);
    l += __shfl_xor(l, 2, 16);
    l += __shfl_xor(l, 4, 16);
    l += __shfl_xor(l, 8, 16);
    lrun[r] = l;
  }
  const int srow0 = sblk * 128 + wave * 16 + half * 8;
#pragma unroll
  for (int jj = 0; jj < 8; ++jj) {
    const int col = h * HDIM + jj * 16 + l16;
#pragma unroll
    for (int r = 0; r < 8; ++r)
      out[(size_t)(srow0 + r) * (NHEAD * HDIM) + col] = O[jj][r] / lrun[r];
  }
}

// ---------------------------------------------------------------------------
extern "C" void kernel_launch(void* const* d_in, const int* in_sizes, int n_in,
                              void* d_out, int out_size, void* d_ws, size_t ws_size,
                              hipStream_t stream) {
  (void)in_sizes; (void)n_in; (void)out_size; (void)ws_size;

  const float*         X    = (const float*)d_in[0];
  const unsigned char* mask = (const unsigned char*)d_in[1];  // jnp bool: 1B/elt
  const float*         Wq   = (const float*)d_in[2];
  const float*         bq   = (const float*)d_in[3];
  const float*         Wk   = (const float*)d_in[4];
  const float*         bk   = (const float*)d_in[5];
  const float*         Wv   = (const float*)d_in[6];
  const float*         bv   = (const float*)d_in[7];
  float* out = (float*)d_out;

  // Workspace: Xbf 4MB | WT 12MB | Q 8MB | K 8MB | Vt 8MB  (~40MB)
  bf16_t* Xbf = (bf16_t*)d_ws;
  bf16_t* WT  = Xbf + (size_t)SEQ * DMODEL;
  bf16_t* Qb  = WT + (size_t)3 * 2048 * DMODEL;
  bf16_t* Kb  = Qb + (size_t)NHEAD * SEQ * HDIM;
  bf16_t* Vtb = Kb + (size_t)NHEAD * SEQ * HDIM;

  convert_x_kernel<<<dim3((SEQ * DMODEL) / (256 * 4)), 256, 0, stream>>>(X, Xbf);
  transpose_w_kernel<<<dim3(16, 32, 3), 256, 0, stream>>>(Wq, Wk, Wv, WT);
  qkv_gemm_kernel<<<dim3(48, 16), 256, 0, stream>>>(Xbf, WT, bq, bk, bv, Qb, Kb, Vtb);
  attn_kernel<<<dim3(16, 16), 256, 0, stream>>>(Qb, Kb, Vtb, mask, out);
}
